// upsample_edgeConv_23691039605457
// MI455X (gfx1250) — matrix operand order repaired and checked by reading, not compiled, not run on hardware
//
#include <hip/hip_runtime.h>

typedef __attribute__((ext_vector_type(16))) _Float16 v16h;
typedef __attribute__((ext_vector_type(8)))  _Float16 v8h;
typedef __attribute__((ext_vector_type(8)))  float    v8f;

#define BB    8
#define FIN   64
#define NP    2048
#define KNN   20
#define CCH   128          // 2*FIN
#define CO1   256          // 4*FIN
#define TAP1  11           // K/2+1
#define KW1   1408         // CCH*TAP1
#define J1    10           // KNN - TAP1 + 1
#define COLS1 (NP*J1)      // 20480
#define CNT1  (BB*COLS1)   // 163840
#define CO2   256          // 2*FOUT
#define TAP2  40           // 2*K
#define KW2   5120         // CCH*TAP2
#define CNT2  (BB*NP)      // 16384
#define HFLAT 2560         // KNN*CCH == CO1*J1, per (b,n)

#define WMMA16(A, Bf, C) \
  __builtin_amdgcn_wmma_f32_16x16x32_f16(false, (A), false, (Bf), (short)0, (C), false, false)

// load a 16x32 f16 A/B fragment: 16 contiguous f16 per lane as two b128 loads
__device__ __forceinline__ v16h load_frag(const _Float16* __restrict__ p) {
  v8h lo = *(const v8h*)p;
  v8h hi = *(const v8h*)(p + 8);
  v16h f;
#pragma unroll
  for (int e = 0; e < 8; ++e) { f[e] = lo[e]; f[e + 8] = hi[e]; }
  return f;
}
// A fragment: per-lane K is contiguous {0..15} (+16 for upper lanes)
__device__ __forceinline__ v16h load_frag_a(const _Float16* __restrict__ p) {
  v8h lo = *(const v8h*)p;
  v8h hi = *(const v8h*)(p + 8);
  v16h f;
#pragma unroll
  for (int e = 0; e < 8; ++e) { f[e] = lo[e]; f[e + 8] = hi[e]; }
  return f;
}

// ---------------------------------------------------------------- weight permute+cvt
// K-axis reordered as k = t*CCH + ci (tap-major, channel-minor)
__global__ void k_perm_w1(const float* __restrict__ w, _Float16* __restrict__ wp) {
  int i = blockIdx.x * blockDim.x + threadIdx.x;
  if (i >= CO1 * KW1) return;
  int co = i / KW1; int k = i - co * KW1;
  int t = k >> 7; int ci = k & 127;
  wp[i] = (_Float16)w[(co * CCH + ci) * TAP1 + t];
}

__global__ void k_perm_w2(const float* __restrict__ w, _Float16* __restrict__ wp) {
  int i = blockIdx.x * blockDim.x + threadIdx.x;
  if (i >= CO2 * KW2) return;
  int co = i / KW2; int k = i - co * KW2;
  int s = k >> 7; int ci = k & 127;
  wp[i] = (_Float16)w[(co * CCH + ci) * TAP2 + s];
}

__global__ void k_sqnorm(const float* __restrict__ x, float* __restrict__ sq) {
  int i = blockIdx.x * blockDim.x + threadIdx.x;
  if (i >= BB * NP) return;
  int b = i / NP, n = i - b * NP;
  const float* xp = x + (size_t)b * FIN * NP + n;
  float s = 0.f;
#pragma unroll
  for (int d = 0; d < FIN; ++d) { float v = xp[d * NP]; s += v * v; }
  sq[i] = s;
}

// ---------------------------------------------------------------- kNN top-21
__global__ __launch_bounds__(256) void k_knn(const float* __restrict__ x,
                                             const float* __restrict__ sq,
                                             int* __restrict__ idx) {
  __shared__ float xs[FIN * 128];
  __shared__ float sqs[128];
  const int tid = threadIdx.x;
  const int b = blockIdx.x >> 3;                 // NP/256 == 8 blocks per batch
  const int n = ((blockIdx.x & 7) << 8) + tid;
  const float* xb = x + (size_t)b * FIN * NP;

  float xn[FIN];
#pragma unroll
  for (int d = 0; d < FIN; ++d) xn[d] = xb[d * NP + n];
  const float sqn = sq[b * NP + n];

  float td[KNN + 1]; int ti[KNN + 1];
#pragma unroll
  for (int q = 0; q <= KNN; ++q) { td[q] = 3.4e38f; ti[q] = 0; }

  for (int m0 = 0; m0 < NP; m0 += 128) {
    __syncthreads();
    for (int i = 0; i < (FIN * 128) / 256; ++i) {
      int flat = i * 256 + tid;
      xs[flat] = xb[(flat >> 7) * NP + m0 + (flat & 127)];
    }
    if (tid < 128) sqs[tid] = sq[b * NP + m0 + tid];
    __syncthreads();
    for (int mm = 0; mm < 128; ++mm) {
      float dot = 0.f;
#pragma unroll
      for (int d = 0; d < FIN; ++d) dot = fmaf(xn[d], xs[d * 128 + mm], dot);
      float dist = sqn + sqs[mm] - 2.f * dot;
      if (dist < td[KNN]) {                      // insert + single bubble pass
        td[KNN] = dist; ti[KNN] = m0 + mm;
#pragma unroll
        for (int s = KNN; s > 0; --s) {
          if (td[s] < td[s - 1]) {
            float tf = td[s]; td[s] = td[s - 1]; td[s - 1] = tf;
            int tt = ti[s]; ti[s] = ti[s - 1]; ti[s - 1] = tt;
          }
        }
      }
    }
  }
  int* op = idx + ((size_t)b * NP + n) * KNN;    // drop slot 0 (self)
#pragma unroll
  for (int q = 0; q < KNN; ++q) op[q] = ti[q + 1];
}

// ---------------------------------------------------------------- edge features -> f16, channel-last
// eeT[b][n][kk][ci]  (contiguous in ci -> B-matrix rows become contiguous streams)
__global__ void k_edge(const float* __restrict__ x, const int* __restrict__ idx,
                       _Float16* __restrict__ eeT) {
  int i = blockIdx.x * blockDim.x + threadIdx.x;
  if (i >= BB * NP * KNN * CCH) return;
  int ci = i & 127;
  int r  = i >> 7;            // (b, n, kk)
  int kk = r % KNN;
  int r2 = r / KNN;
  int n  = r2 & (NP - 1);
  int b  = r2 >> 11;
  float v;
  if (ci < FIN) {
    v = x[((size_t)b * FIN + ci) * NP + n];                       // central (broadcast over kk)
  } else {
    int c = ci - FIN;
    int m = idx[((size_t)b * NP + n) * KNN + kk];
    const float* xc = x + ((size_t)b * FIN + c) * NP;
    v = xc[m] - xc[n];                                            // neighbor - central
  }
  eeT[i] = (_Float16)v;
}

// ---------------------------------------------------------------- conv1: 32x32 register-blocked WMMA GEMM
// out[co][b][c] = sum_k w1p[co][k] * eeT-row(c),  c = n*10+j, k = t*128+ci (contiguous streams)
__global__ __launch_bounds__(128) void k_conv1_wmma(const _Float16* __restrict__ w1p,
                                                    const _Float16* __restrict__ eeT,
                                                    const float* __restrict__ bias,
                                                    float* __restrict__ hraw) {
  const int lane  = threadIdx.x & 31;
  const int gid   = blockIdx.x * 4 + (threadIdx.x >> 5);
  const int tileM = gid & 7;                     // 8 supertiles of 32 rows
  const int rest  = gid >> 3;
  const int colT  = rest % (COLS1 / 32);         // 640 supertiles of 32 cols
  const int b     = rest / (COLS1 / 32);
  const int lo    = lane & 15;
  const int hi    = lane >> 4;
  const int co0   = tileM * 32 + lo;
  const int c0    = colT * 32 + lo;
  const int c1    = c0 + 16;
  const int n0 = c0 / J1, j0 = c0 - n0 * J1;
  const int n1 = c1 / J1, j1 = c1 - n1 * J1;
  const _Float16* ap0 = w1p + (size_t)co0 * KW1 + hi * 16;
  const _Float16* ap1 = ap0 + (size_t)16 * KW1;
  const _Float16* bp0 = eeT + ((size_t)(b * NP + n0) * KNN + j0) * CCH + hi * 16;
  const _Float16* bp1 = eeT + ((size_t)(b * NP + n1) * KNN + j1) * CCH + hi * 16;
  v8f acc00 = {}, acc01 = {}, acc10 = {}, acc11 = {};
  for (int k0 = 0; k0 < KW1; k0 += 32) {
    v16h A0 = load_frag_a(ap0 + k0);
    v16h A1 = load_frag_a(ap1 + k0);
    v16h B0 = load_frag(bp0 + k0);
    v16h B1 = load_frag(bp1 + k0);
    __builtin_prefetch(bp0 + k0 + 1024, 0, 1);
    __builtin_prefetch(bp1 + k0 + 1024, 0, 1);
    acc00 = WMMA16(A0, B0, acc00);
    acc01 = WMMA16(A0, B1, acc01);
    acc10 = WMMA16(A1, B0, acc10);
    acc11 = WMMA16(A1, B1, acc11);
  }
#pragma unroll
  for (int r = 0; r < 8; ++r) {
    int cor0 = tileM * 32 + hi * 8 + r;
    int cor1 = cor0 + 16;
    float bi0 = bias[cor0], bi1 = bias[cor1];
    hraw[(size_t)cor0 * CNT1 + (size_t)b * COLS1 + c0] = acc00[r] + bi0;
    hraw[(size_t)cor0 * CNT1 + (size_t)b * COLS1 + c1] = acc01[r] + bi0;
    hraw[(size_t)cor1 * CNT1 + (size_t)b * COLS1 + c0] = acc10[r] + bi1;
    hraw[(size_t)cor1 * CNT1 + (size_t)b * COLS1 + c1] = acc11[r] + bi1;
  }
}

// ---------------------------------------------------------------- BN stats (per channel)
__global__ __launch_bounds__(256) void k_stats(const float* __restrict__ src, int cnt,
                                               float* __restrict__ mean,
                                               float* __restrict__ inv) {
  __shared__ float sh[256], sh2[256];
  int ch = blockIdx.x, tid = threadIdx.x;
  const float* p = src + (size_t)ch * cnt;
  float s = 0.f, s2 = 0.f;
  for (int i = tid; i < cnt; i += 256) { float v = p[i]; s += v; s2 += v * v; }
  sh[tid] = s; sh2[tid] = s2; __syncthreads();
  for (int o = 128; o > 0; o >>= 1) {
    if (tid < o) { sh[tid] += sh[tid + o]; sh2[tid] += sh2[tid + o]; }
    __syncthreads();
  }
  if (tid == 0) {
    float m = sh[0] / (float)cnt;
    float v = sh2[0] / (float)cnt - m * m;
    mean[ch] = m; inv[ch] = rsqrtf(v + 1e-5f);
  }
}

// ---------------------------------------------------------------- BN1 + LeakyReLU -> hT[b][n][kk][ci]
// inte[b][ci][n][kk] = h[b][co][n][j] with ci = (co*10+j)/20, kk = (co*10+j)%20 (channel-last store)
__global__ void k_bn1(const float* __restrict__ hraw, const float* __restrict__ mean,
                      const float* __restrict__ inv, const float* __restrict__ g,
                      const float* __restrict__ be, _Float16* __restrict__ hT) {
  int i = blockIdx.x * blockDim.x + threadIdx.x;
  if (i >= CO1 * CNT1) return;
  int co = i / CNT1; int bc = i - co * CNT1;
  float v = (hraw[i] - mean[co]) * inv[co] * g[co] + be[co];
  v = v > 0.f ? v : 0.01f * v;
  int b = bc / COLS1; int c = bc - b * COLS1;
  int n = c / J1; int j = c - n * J1;
  int pos = co * J1 + j;               // 0..2559 within the (b,n) flat vector
  int ci = pos / KNN; int kk = pos - ci * KNN;
  hT[(size_t)(b * NP + n) * HFLAT + kk * CCH + ci] = (_Float16)v;
}

// ---------------------------------------------------------------- conv2: 32x32 register-blocked WMMA GEMM
// merge K-axis k = s*128+ci: k<2560 -> eeT stream, k>=2560 -> hT stream (both contiguous)
__global__ __launch_bounds__(128) void k_conv2_wmma(const _Float16* __restrict__ w2p,
                                                    const _Float16* __restrict__ eeT,
                                                    const _Float16* __restrict__ hT,
                                                    const float* __restrict__ bias,
                                                    float* __restrict__ craw) {
  const int lane  = threadIdx.x & 31;
  const int gid   = blockIdx.x * 4 + (threadIdx.x >> 5);
  const int tileM = gid & 7;                     // 8 supertiles of 32 rows
  const int rest  = gid >> 3;
  const int colT  = rest % (NP / 32);            // 64 supertiles of 32 cols
  const int b     = rest / (NP / 32);
  const int lo    = lane & 15;
  const int hi    = lane >> 4;
  const int co0   = tileM * 32 + lo;
  const int n0    = colT * 32 + lo;
  const int n1    = n0 + 16;
  const _Float16* ap0  = w2p + (size_t)co0 * KW2 + hi * 16;
  const _Float16* ap1  = ap0 + (size_t)16 * KW2;
  const _Float16* ebp0 = eeT + (size_t)(b * NP + n0) * HFLAT + hi * 16;
  const _Float16* ebp1 = eeT + (size_t)(b * NP + n1) * HFLAT + hi * 16;
  const _Float16* hbp0 = hT  + (size_t)(b * NP + n0) * HFLAT + hi * 16;
  const _Float16* hbp1 = hT  + (size_t)(b * NP + n1) * HFLAT + hi * 16;
  v8f acc00 = {}, acc01 = {}, acc10 = {}, acc11 = {};
  for (int k0 = 0; k0 < HFLAT; k0 += 32) {                 // first half: ee part
    v16h A0 = load_frag_a(ap0 + k0);
    v16h A1 = load_frag_a(ap1 + k0);
    v16h B0 = load_frag(ebp0 + k0);
    v16h B1 = load_frag(ebp1 + k0);
    __builtin_prefetch(ebp0 + k0 + 1024, 0, 1);
    __builtin_prefetch(ebp1 + k0 + 1024, 0, 1);
    acc00 = WMMA16(A0, B0, acc00);
    acc01 = WMMA16(A0, B1, acc01);
    acc10 = WMMA16(A1, B0, acc10);
    acc11 = WMMA16(A1, B1, acc11);
  }
  for (int k0 = 0; k0 < HFLAT; k0 += 32) {                 // second half: inte part
    v16h A0 = load_frag_a(ap0 + HFLAT + k0);
    v16h A1 = load_frag_a(ap1 + HFLAT + k0);
    v16h B0 = load_frag(hbp0 + k0);
    v16h B1 = load_frag(hbp1 + k0);
    __builtin_prefetch(hbp0 + k0 + 1024, 0, 1);
    __builtin_prefetch(hbp1 + k0 + 1024, 0, 1);
    acc00 = WMMA16(A0, B0, acc00);
    acc01 = WMMA16(A0, B1, acc01);
    acc10 = WMMA16(A1, B0, acc10);
    acc11 = WMMA16(A1, B1, acc11);
  }
#pragma unroll
  for (int r = 0; r < 8; ++r) {
    int cor0 = tileM * 32 + hi * 8 + r;
    int cor1 = cor0 + 16;
    float bi0 = bias[cor0], bi1 = bias[cor1];
    craw[(size_t)cor0 * CNT2 + (size_t)b * NP + n0] = acc00[r] + bi0;
    craw[(size_t)cor0 * CNT2 + (size_t)b * NP + n1] = acc01[r] + bi0;
    craw[(size_t)cor1 * CNT2 + (size_t)b * NP + n0] = acc10[r] + bi1;
    craw[(size_t)cor1 * CNT2 + (size_t)b * NP + n1] = acc11[r] + bi1;
  }
}

// ---------------------------------------------------------------- BN2 + ReLU + final reshape (B,128,4096)
__global__ void k_out(const float* __restrict__ craw, const float* __restrict__ mean,
                      const float* __restrict__ inv, const float* __restrict__ g,
                      const float* __restrict__ be, float* __restrict__ out) {
  int i = blockIdx.x * blockDim.x + threadIdx.x;
  if (i >= BB * 128 * 4096) return;
  int b = i / (128 * 4096); int r = i - b * (128 * 4096);
  int f = r / 4096; int p = r - f * 4096;
  int co = f * 2 + (p >> 11);
  int n  = p & 2047;
  float v = craw[(size_t)co * CNT2 + (size_t)b * NP + n];
  v = (v - mean[co]) * inv[co] * g[co] + be[co];
  out[i] = v > 0.f ? v : 0.f;
}

// ---------------------------------------------------------------- host
extern "C" void kernel_launch(void* const* d_in, const int* in_sizes, int n_in,
                              void* d_out, int out_size, void* d_ws, size_t ws_size,
                              hipStream_t stream) {
  (void)in_sizes; (void)n_in; (void)out_size; (void)ws_size;
  const float* x   = (const float*)d_in[0];
  const float* w1  = (const float*)d_in[1];
  const float* b1  = (const float*)d_in[2];
  const float* g1  = (const float*)d_in[3];
  const float* be1 = (const float*)d_in[4];
  const float* w2  = (const float*)d_in[5];
  const float* b2  = (const float*)d_in[6];
  const float* g2  = (const float*)d_in[7];
  const float* be2 = (const float*)d_in[8];

  char* ws = (char*)d_ws; size_t off = 0;
  auto alloc = [&](size_t bytes) -> void* {
    void* p = ws + off; off += bytes; off = (off + 255) & ~(size_t)255; return p;
  };
  float*    sq    = (float*)alloc((size_t)BB * NP * sizeof(float));
  int*      idx   = (int*)alloc((size_t)BB * NP * KNN * sizeof(int));
  _Float16* w1p   = (_Float16*)alloc((size_t)CO1 * KW1 * 2);
  _Float16* w2p   = (_Float16*)alloc((size_t)CO2 * KW2 * 2);
  _Float16* eeT   = (_Float16*)alloc((size_t)BB * NP * KNN * CCH * 2);
  float*    hraw  = (float*)alloc((size_t)CO1 * CNT1 * 4);
  _Float16* hT    = (_Float16*)alloc((size_t)BB * NP * HFLAT * 2);
  float*    craw  = (float*)alloc((size_t)CO2 * CNT2 * 4);
  float*    mean1 = (float*)alloc(CO1 * 4);
  float*    inv1  = (float*)alloc(CO1 * 4);
  float*    mean2 = (float*)alloc(CO2 * 4);
  float*    inv2  = (float*)alloc(CO2 * 4);

  k_perm_w1<<<(CO1 * KW1 + 255) / 256, 256, 0, stream>>>(w1, w1p);
  k_perm_w2<<<(CO2 * KW2 + 255) / 256, 256, 0, stream>>>(w2, w2p);
  k_sqnorm<<<(BB * NP + 255) / 256, 256, 0, stream>>>(x, sq);
  k_knn<<<BB * (NP / 256), 256, 0, stream>>>(x, sq, idx);
  k_edge<<<(BB * NP * KNN * CCH + 255) / 256, 256, 0, stream>>>(x, idx, eeT);
  k_conv1_wmma<<<(8 * (COLS1 / 32) * BB) / 4, 128, 0, stream>>>(w1p, eeT, b1, hraw);
  k_stats<<<CO1, 256, 0, stream>>>(hraw, CNT1, mean1, inv1);
  k_bn1<<<(CO1 * CNT1 + 255) / 256, 256, 0, stream>>>(hraw, mean1, inv1, g1, be1, hT);
  k_conv2_wmma<<<(8 * (NP / 32) * BB) / 4, 128, 0, stream>>>(w2p, eeT, hT, b2, craw);
  k_stats<<<CO2, 256, 0, stream>>>(craw, CNT2, mean2, inv2);
  k_out<<<(BB * 128 * 4096 + 255) / 256, 256, 0, stream>>>(craw, mean2, inv2, g2, be2,
                                                           (float*)d_out);
}
